// SplitHeadSAMEncoderAttention_44212393345684
// MI455X (gfx1250) — compile-verified, matrix operand order checked by reading, not run on hardware
//
#include <hip/hip_runtime.h>

typedef __attribute__((ext_vector_type(16))) _Float16 v16h;
typedef __attribute__((ext_vector_type(8)))  _Float16 v8h;
typedef __attribute__((ext_vector_type(8)))  float    v8f;

static constexpr int kHW = 4096;   // 64*64 tokens
static constexpr int kC  = 768;    // channels = nH*hd
static constexpr int kNH = 12;
static constexpr int kHD = 64;
static constexpr int kSZ = 64;     // H == W == 64

static __device__ __forceinline__ int laneid() { return threadIdx.x & 31; }

static __device__ __forceinline__ v8f zero8() {
  v8f z;
#pragma unroll
  for (int i = 0; i < 8; ++i) z[i] = 0.0f;
  return z;
}

// ---- CDNA5 async memory->LDS copy (ASYNCcnt-tracked, no VGPR round trip) ---
// Per-lane 16B: LDS[vdst_lane] = MEM[vaddr_lane]. Generic LDS pointer low 32
// bits == LDS offset (aperture form), valid as the VDST LDS address.
static __device__ __forceinline__ void async_cp16(const void* g, void* l) {
  asm volatile("global_load_async_to_lds_b128 %0, %1, off"
               :: "v"((unsigned)(uintptr_t)l), "v"(g)
               : "memory");
}
static __device__ __forceinline__ void wait_async() {
  asm volatile("s_wait_asynccnt 0x0" ::: "memory");
}

// ---- WMMA fragment loaders (CDNA5 16x16x32 f16, wave32) -------------------
// A-matrix 16x32 layout (ISA 7.12.2): lanes 0-15 = rows 0-15 holding K 0..7 /
// 16..23 in elems 0..7 / 8..15; lanes 16-31 = same rows holding K 8..15 / 24..31.
static __device__ __forceinline__ v16h frag_row(const _Float16* p, int ld) {
  const int l = laneid();
  const _Float16* s = p + (l & 15) * ld;
  const int hh = (l >> 4) * 8;
  v16h f;
#pragma unroll
  for (int i = 0; i < 8; ++i) f[i] = s[hh + i];
#pragma unroll
  for (int i = 0; i < 8; ++i) f[8 + i] = s[16 + hh + i];
  return f;
}

// B-matrix from a K-major (K x N row-major) source: lane's (l&15) is the column.
static __device__ __forceinline__ v16h frag_col(const _Float16* p, int ld) {
  const int l = laneid();
  const int c = l & 15;
  const int hh = (l >> 4) * 8;
  v16h f;
#pragma unroll
  for (int i = 0; i < 8; ++i) f[i] = p[(hh + i) * ld + c];
#pragma unroll
  for (int i = 0; i < 8; ++i) f[8 + i] = p[(16 + hh + i) * ld + c];
  return f;
}

// B-fragment gathered from rel_pos (f32), col = bias index k, row = d.
// B[d][k] = 8 * rel_pos[(qpos - k + 63)][d] ; the x8 undoes the softmax scale
// folded into Q so that  (q*s) . (R/s) == q . R.
static __device__ __forceinline__ v16h frag_rel(const float* rp, int qpos, int ncol0, int kc) {
  const int l = laneid();
  const int c = ncol0 + (l & 15);
  const int hh = (l >> 4) * 8;
  const float* s = rp + (size_t)(qpos - c + 63) * kHD + kc * 32;
  v16h f;
#pragma unroll
  for (int i = 0; i < 8; ++i) f[i] = (_Float16)(s[hh + i] * 8.0f);
#pragma unroll
  for (int i = 0; i < 8; ++i) f[8 + i] = (_Float16)(s[16 + hh + i] * 8.0f);
  return f;
}

static __device__ __forceinline__ v8f wmma16(v16h a, v16h b, v8f c) {
  return __builtin_amdgcn_wmma_f32_16x16x32_f16(false, a, false, b, (short)0, c, false, false);
}

// ---------------------------------------------------------------------------
__global__ void cvt_f32_to_f16(const float* __restrict__ src, _Float16* __restrict__ dst, int n) {
  int i = blockIdx.x * blockDim.x + threadIdx.x;
  if (i < n) dst[i] = (_Float16)src[i];
}

// ---- QKV projection: (4096 x 768) x (768 x 768)^T + bias, out head-major f16
// Q output is pre-scaled by hd^-0.5 = 0.125 (softmax scale folded in).
#define LDS_LD 40   // 32 + 8 halves pad (rows stay 16B aligned)
__global__ __launch_bounds__(256) void gemm_qkv(
    const _Float16* __restrict__ Xh,
    const _Float16* __restrict__ Wq, const _Float16* __restrict__ Wk, const _Float16* __restrict__ Wv,
    const float* __restrict__ bq, const float* __restrict__ bk, const float* __restrict__ bv,
    _Float16* __restrict__ Qh, _Float16* __restrict__ Kh, _Float16* __restrict__ Vh) {
  const int which = blockIdx.z;
  const _Float16* W = (which == 0) ? Wq : (which == 1) ? Wk : Wv;
  const float* bias = (which == 0) ? bq : (which == 1) ? bk : bv;
  _Float16* Out     = (which == 0) ? Qh : (which == 1) ? Kh : Vh;
  const float sc    = (which == 0) ? 0.125f : 1.0f;

  const int m0 = blockIdx.y * 128;
  const int n0 = blockIdx.x * 64;
  __shared__ _Float16 sA[2][128 * LDS_LD];
  __shared__ _Float16 sB[2][64 * LDS_LD];

  const int tid = threadIdx.x;
  const int wv_ = tid >> 5;
  const int wm = wv_ & 3;        // 4 waves over M (32 rows each)
  const int wn = wv_ >> 2;       // 2 waves over N (32 cols each)

  auto issue = [&](int k0, int b) {
#pragma unroll
    for (int c = tid; c < 512; c += 256) {      // A: 128x32 halves
      int row = c >> 2, cc = (c & 3) * 8;
      async_cp16(&Xh[(size_t)(m0 + row) * kC + k0 + cc], &sA[b][row * LDS_LD + cc]);
    }
    {                                           // B: 64x32 halves (o-major)
      int row = tid >> 2, cc = (tid & 3) * 8;
      async_cp16(&W[(size_t)(n0 + row) * kC + k0 + cc], &sB[b][row * LDS_LD + cc]);
    }
  };

  v8f acc[2][2];
#pragma unroll
  for (int a = 0; a < 2; ++a)
#pragma unroll
    for (int b = 0; b < 2; ++b) acc[a][b] = zero8();

  issue(0, 0);
  wait_async();
  __syncthreads();

  for (int kt = 0; kt < 24; ++kt) {             // 768 / 32
    const int cur = kt & 1;
    if (kt + 1 < 24) issue((kt + 1) * 32, cur ^ 1);   // prefetch next tile

    v16h af[2], bf[2];
#pragma unroll
    for (int mi = 0; mi < 2; ++mi) af[mi] = frag_row(&sA[cur][(wm * 32 + mi * 16) * LDS_LD], LDS_LD);
#pragma unroll
    for (int ni = 0; ni < 2; ++ni) bf[ni] = frag_row(&sB[cur][(wn * 32 + ni * 16) * LDS_LD], LDS_LD);
#pragma unroll
    for (int mi = 0; mi < 2; ++mi)
#pragma unroll
      for (int ni = 0; ni < 2; ++ni) acc[mi][ni] = wmma16(af[mi], bf[ni], acc[mi][ni]);

    wait_async();
    __syncthreads();
  }

  const int l = tid & 31, col = l & 15, rbase = (l >> 4) * 8;
#pragma unroll
  for (int mi = 0; mi < 2; ++mi)
#pragma unroll
    for (int ni = 0; ni < 2; ++ni) {
      int o = n0 + wn * 32 + ni * 16 + col;
      int head = o >> 6, d = o & 63;
      float bo = bias[o];
#pragma unroll
      for (int v = 0; v < 8; ++v) {
        int m = m0 + wm * 32 + mi * 16 + rbase + v;
        Out[(size_t)head * kHW * kHD + (size_t)m * kHD + d] = (_Float16)((acc[mi][ni][v] + bo) * sc);
      }
    }
}

// ---- rel_h / rel_w bias GEMMs: per (head, fixed row/col): 64x64x64 ---------
__global__ __launch_bounds__(128) void rel_bias(
    const _Float16* __restrict__ Qh, const float* __restrict__ rph, const float* __restrict__ rpw,
    float* __restrict__ relH, float* __restrict__ relW) {
  const int mode = blockIdx.y;              // 0: rel_h (fixed h), 1: rel_w (fixed w)
  const int n = blockIdx.x >> 6;
  const int p = blockIdx.x & 63;
  const float* rp = mode ? rpw : rph;
  float* outp = mode ? relW : relH;

  const size_t base = (size_t)n * kHW * kHD + (mode == 0 ? (size_t)p * kSZ * kHD : (size_t)p * kHD);
  const int rowStride = (mode == 0) ? kHD : kSZ * kHD;

  const int wv_ = threadIdx.x >> 5;
  const int l = threadIdx.x & 31, col = l & 15, rbase = (l >> 4) * 8;

  const _Float16* Ab = Qh + base + (size_t)(wv_ * 16) * rowStride;
  v16h a0 = frag_row(Ab, rowStride);        // K (=d) 0..31
  v16h a1 = frag_row(Ab + 32, rowStride);   // K 32..63

  v8f acc[4];
#pragma unroll
  for (int nt = 0; nt < 4; ++nt) acc[nt] = zero8();
#pragma unroll
  for (int nt = 0; nt < 4; ++nt) {
    acc[nt] = wmma16(a0, frag_rel(rp, p, nt * 16, 0), acc[nt]);
    acc[nt] = wmma16(a1, frag_rel(rp, p, nt * 16, 1), acc[nt]);
  }

  float* ob = outp + base;
#pragma unroll
  for (int nt = 0; nt < 4; ++nt)
#pragma unroll
    for (int v = 0; v < 8; ++v)
      ob[(size_t)(wv_ * 16 + rbase + v) * rowStride + nt * 16 + col] = acc[nt][v];
}

// ---- Flash attention: one block per (head, 64-query tile), 4 waves --------
// Double-buffered async K/V tiles; rel_h in LDS; rel_w held in registers and
// fed through the WMMA C operand together with rel_h (bias adds are free).
#define KV_LD 72    // 64 + 8 pad
__global__ __launch_bounds__(128) void flash_attn(
    const _Float16* __restrict__ Qh, const _Float16* __restrict__ Kh, const _Float16* __restrict__ Vh,
    const float* __restrict__ relH, const float* __restrict__ relW, _Float16* __restrict__ AO) {
  const int n = blockIdx.y;
  const int q0 = blockIdx.x * 64;

  __shared__ _Float16 sK[2][64 * KV_LD];
  __shared__ _Float16 sV[2][64 * KV_LD];
  __shared__ float sRH[64 * 64];
  __shared__ _Float16 sP[4][16 * KV_LD];

  const int tid = threadIdx.x;
  const int wv_ = tid >> 5;
  const int l = tid & 31, col = l & 15, rbase = (l >> 4) * 8;

  const size_t hb = ((size_t)n * kHW + q0) * kHD;

  auto issueKV = [&](int kb, int b) {
    const size_t kvb = ((size_t)n * kHW + kb * 64) * kHD;
#pragma unroll
    for (int c = tid; c < 512; c += 128) {
      int row = c >> 3, cc = (c & 7) * 8;
      async_cp16(&Kh[kvb + (size_t)row * kHD + cc], &sK[b][row * KV_LD + cc]);
      async_cp16(&Vh[kvb + (size_t)row * kHD + cc], &sV[b][row * KV_LD + cc]);
    }
  };

  // stage rel_h tile (64 q x 64 kh, f32) and first K/V tile asynchronously
#pragma unroll
  for (int c = tid; c < 1024; c += 128)
    async_cp16(relH + hb + c * 4, &sRH[c * 4]);
  issueKV(0, 0);

  // while those fly: rel_w per-lane constants and Q fragments into registers
  float rw[4][8];
  {
    const float* rwg = relW + hb;
#pragma unroll
    for (int nt = 0; nt < 4; ++nt)
#pragma unroll
      for (int v = 0; v < 8; ++v)
        rw[nt][v] = rwg[(size_t)(wv_ * 16 + rbase + v) * 64 + nt * 16 + col];
  }
  const _Float16* qp = Qh + hb + (size_t)(wv_ * 16) * kHD;
  v16h aQ0 = frag_row(qp, kHD);
  v16h aQ1 = frag_row(qp + 32, kHD);

  float mrow[8], lrow[8];
#pragma unroll
  for (int v = 0; v < 8; ++v) { mrow[v] = -1e30f; lrow[v] = 0.0f; }
  v8f oacc[4];
#pragma unroll
  for (int dt = 0; dt < 4; ++dt) oacc[dt] = zero8();

  wait_async();
  __syncthreads();

  for (int kb = 0; kb < 64; ++kb) {           // key block == one image row (kh = kb)
    const int cur = kb & 1;
    if (kb + 1 < 64) issueKV(kb + 1, cur ^ 1);   // prefetch next tile

    // S = Q K^T + bias, bias seeded through the WMMA C operand
    v8f sac[4];
#pragma unroll
    for (int v = 0; v < 8; ++v) {
      float bh = sRH[(wv_ * 16 + rbase + v) * 64 + kb];
#pragma unroll
      for (int nt = 0; nt < 4; ++nt) sac[nt][v] = bh + rw[nt][v];
    }
#pragma unroll
    for (int nt = 0; nt < 4; ++nt) {
      sac[nt] = wmma16(aQ0, frag_row(&sK[cur][(nt * 16) * KV_LD], KV_LD), sac[nt]);
      sac[nt] = wmma16(aQ1, frag_row(&sK[cur][(nt * 16) * KV_LD + 32], KV_LD), sac[nt]);
    }

    // online softmax (row reductions across the 16 lanes of each half)
    float mnew[8], corr[8];
#pragma unroll
    for (int v = 0; v < 8; ++v) {
      float mx = fmaxf(fmaxf(sac[0][v], sac[1][v]), fmaxf(sac[2][v], sac[3][v]));
#pragma unroll
      for (int off = 1; off < 16; off <<= 1) mx = fmaxf(mx, __shfl_xor(mx, off));
      mnew[v] = fmaxf(mrow[v], mx);
      corr[v] = __expf(mrow[v] - mnew[v]);
      mrow[v] = mnew[v];
    }
#pragma unroll
    for (int v = 0; v < 8; ++v) {
      float psum = 0.0f;
#pragma unroll
      for (int nt = 0; nt < 4; ++nt) {
        float pe = __expf(sac[nt][v] - mnew[v]);
        psum += pe;
        sP[wv_][(rbase + v) * KV_LD + nt * 16 + col] = (_Float16)pe;
      }
#pragma unroll
      for (int off = 1; off < 16; off <<= 1) psum += __shfl_xor(psum, off);
      lrow[v] = lrow[v] * corr[v] + psum;
#pragma unroll
      for (int dt = 0; dt < 4; ++dt) oacc[dt][v] *= corr[v];
    }

    // O += P V  (wave-private P staged through LDS to re-fragment)
    const _Float16* pp = sP[wv_];
    v16h aP0 = frag_row(pp, KV_LD);
    v16h aP1 = frag_row(pp + 32, KV_LD);
#pragma unroll
    for (int dt = 0; dt < 4; ++dt) {
      oacc[dt] = wmma16(aP0, frag_col(&sV[cur][dt * 16], KV_LD), oacc[dt]);
      oacc[dt] = wmma16(aP1, frag_col(&sV[cur][32 * KV_LD + dt * 16], KV_LD), oacc[dt]);
    }

    wait_async();      // next tile landed in LDS
    __syncthreads();   // all waves done reading current buffers
  }

  // normalize and write attention output, token-major (m, o) f16
#pragma unroll
  for (int dt = 0; dt < 4; ++dt)
#pragma unroll
    for (int v = 0; v < 8; ++v) {
      int qr = q0 + wv_ * 16 + rbase + v;
      int o = n * kHD + dt * 16 + col;
      AO[(size_t)qr * kC + o] = (_Float16)(oacc[dt][v] / lrow[v]);
    }
}

// ---- Output projection: (4096 x 768) x (768 x 768)^T + bias -> f32 out -----
__global__ __launch_bounds__(256) void gemm_proj(
    const _Float16* __restrict__ Ah, const _Float16* __restrict__ Wh,
    const float* __restrict__ bias, float* __restrict__ out) {
  const int m0 = blockIdx.y * 128;
  const int n0 = blockIdx.x * 64;
  __shared__ _Float16 sA[2][128 * LDS_LD];
  __shared__ _Float16 sB[2][64 * LDS_LD];

  const int tid = threadIdx.x;
  const int wv_ = tid >> 5;
  const int wm = wv_ & 3, wn = wv_ >> 2;

  auto issue = [&](int k0, int b) {
#pragma unroll
    for (int c = tid; c < 512; c += 256) {
      int row = c >> 2, cc = (c & 3) * 8;
      async_cp16(&Ah[(size_t)(m0 + row) * kC + k0 + cc], &sA[b][row * LDS_LD + cc]);
    }
    {
      int row = tid >> 2, cc = (tid & 3) * 8;
      async_cp16(&Wh[(size_t)(n0 + row) * kC + k0 + cc], &sB[b][row * LDS_LD + cc]);
    }
  };

  v8f acc[2][2];
#pragma unroll
  for (int a = 0; a < 2; ++a)
#pragma unroll
    for (int b = 0; b < 2; ++b) acc[a][b] = zero8();

  issue(0, 0);
  wait_async();
  __syncthreads();

  for (int kt = 0; kt < 24; ++kt) {
    const int cur = kt & 1;
    if (kt + 1 < 24) issue((kt + 1) * 32, cur ^ 1);

    v16h af[2], bf[2];
#pragma unroll
    for (int mi = 0; mi < 2; ++mi) af[mi] = frag_row(&sA[cur][(wm * 32 + mi * 16) * LDS_LD], LDS_LD);
#pragma unroll
    for (int ni = 0; ni < 2; ++ni) bf[ni] = frag_row(&sB[cur][(wn * 32 + ni * 16) * LDS_LD], LDS_LD);
#pragma unroll
    for (int mi = 0; mi < 2; ++mi)
#pragma unroll
      for (int ni = 0; ni < 2; ++ni) acc[mi][ni] = wmma16(af[mi], bf[ni], acc[mi][ni]);

    wait_async();
    __syncthreads();
  }

  const int l = tid & 31, col = l & 15, rbase = (l >> 4) * 8;
#pragma unroll
  for (int mi = 0; mi < 2; ++mi)
#pragma unroll
    for (int ni = 0; ni < 2; ++ni) {
      int o = n0 + wn * 32 + ni * 16 + col;
      float bo = bias[o];
#pragma unroll
      for (int v = 0; v < 8; ++v) {
        int m = m0 + wm * 32 + mi * 16 + rbase + v;
        out[(size_t)m * kC + o] = acc[mi][ni][v] + bo;
      }
    }
}

// ---------------------------------------------------------------------------
extern "C" void kernel_launch(void* const* d_in, const int* in_sizes, int n_in,
                              void* d_out, int out_size, void* d_ws, size_t ws_size,
                              hipStream_t stream) {
  const float* x   = (const float*)d_in[0];
  const float* wq  = (const float*)d_in[1];
  const float* bq  = (const float*)d_in[2];
  const float* wk  = (const float*)d_in[3];
  const float* bk  = (const float*)d_in[4];
  const float* wv  = (const float*)d_in[5];
  const float* bv  = (const float*)d_in[6];
  const float* rph = (const float*)d_in[7];
  const float* rpw = (const float*)d_in[8];
  const float* pw  = (const float*)d_in[9];
  const float* pb  = (const float*)d_in[10];
  float* out = (float*)d_out;

  char* ws = (char*)d_ws;
  size_t off = 0;
  auto take = [&](size_t bytes) -> char* {
    char* p = ws + off;
    off += (bytes + 255) & ~(size_t)255;
    return p;
  };
  _Float16* Xh  = (_Float16*)take((size_t)kHW * kC * 2);
  _Float16* Wqh = (_Float16*)take((size_t)kC * kC * 2);
  _Float16* Wkh = (_Float16*)take((size_t)kC * kC * 2);
  _Float16* Wvh = (_Float16*)take((size_t)kC * kC * 2);
  _Float16* Pwh = (_Float16*)take((size_t)kC * kC * 2);
  _Float16* Qf  = (_Float16*)take((size_t)kNH * kHW * kHD * 2);
  _Float16* Kf  = (_Float16*)take((size_t)kNH * kHW * kHD * 2);
  _Float16* Vf  = (_Float16*)take((size_t)kNH * kHW * kHD * 2);
  float*    rH  = (float*)take((size_t)kNH * kHW * kHD * 4);
  float*    rW  = (float*)take((size_t)kNH * kHW * kHD * 4);
  _Float16* AOh = (_Float16*)take((size_t)kHW * kC * 2);

  const int thr = 256;
  cvt_f32_to_f16<<<(kHW * kC + thr - 1) / thr, thr, 0, stream>>>(x, Xh, kHW * kC);
  cvt_f32_to_f16<<<(kC * kC + thr - 1) / thr, thr, 0, stream>>>(wq, Wqh, kC * kC);
  cvt_f32_to_f16<<<(kC * kC + thr - 1) / thr, thr, 0, stream>>>(wk, Wkh, kC * kC);
  cvt_f32_to_f16<<<(kC * kC + thr - 1) / thr, thr, 0, stream>>>(wv, Wvh, kC * kC);
  cvt_f32_to_f16<<<(kC * kC + thr - 1) / thr, thr, 0, stream>>>(pw, Pwh, kC * kC);

  gemm_qkv<<<dim3(kC / 64, kHW / 128, 3), 256, 0, stream>>>(
      Xh, Wqh, Wkh, Wvh, bq, bk, bv, Qf, Kf, Vf);

  rel_bias<<<dim3(kNH * kSZ, 2), 128, 0, stream>>>(Qf, rph, rpw, rH, rW);

  flash_attn<<<dim3(kHW / 64, kNH), 128, 0, stream>>>(Qf, Kf, Vf, rH, rW, AOh);

  gemm_proj<<<dim3(kC / 64, kHW / 128), 256, 0, stream>>>(AOh, Pwh, pb, out);
}
